// WindowAttention_1640677507079
// MI455X (gfx1250) — matrix operand order repaired and checked by reading, not compiled, not run on hardware
//
#include <hip/hip_runtime.h>

// ---------------------------------------------------------------------------
// Swin window attention, fused per-window kernel for gfx1250 (MI455X).
// Matmuls via v_wmma_f32_16x16x32_bf16; weight staging via async load-to-LDS.
// ---------------------------------------------------------------------------

typedef __attribute__((ext_vector_type(16))) __bf16 v16bf;
typedef __attribute__((ext_vector_type(8)))  float  v8f;
typedef int v4i __attribute__((vector_size(16)));

#define WH 7
#define WW 7
#define NT 49          // tokens per window
#define NP 64          // padded tokens
#define C_DIM 128
#define H_NUM 4
#define HD 32
#define NWIN 64        // windows per image (mask count)
#define B_TOT 4096

// LDS layout (bytes). Regions are phase-disjoint and reused.
#define STR_QK   264   // bf16 stride for [64][256] q|k buffer
#define STR_X    136   // bf16 stride for [64][128] x / O buffer
#define STR_WT   48    // bf16 stride for transposed weight panel [384][32]
#define STR_VT   72    // bf16 stride for V^T [4][32][64]
#define STR_S    68    // f32 stride for scores [4][64][64]

#define OFF_QK   0                                 // also sWP in phase D
#define OFF_VT   (OFF_QK + NP*STR_QK*2)            // 33792
#define OFF_X    (OFF_VT + H_NUM*HD*STR_VT*2)      // 52224  (sX, later sO)
#define OFF_WT0  (OFF_X  + NP*STR_X*2)             // 69632  (phase A only)
#define OFF_WT1  (OFF_WT0 + 384*STR_WT*2)          // 106496 (phase A only)
#define OFF_S    OFF_WT0                           // 69632  (phases B/C)
#define SMEM_BYTES (OFF_WT1 + 384*STR_WT*2)        // 143360

// WS layout (bytes)
#define WS_WQKVT  0                // bf16 [384][128] = 98304 B
#define WS_WPROJT 98304            // bf16 [128][128] = 32768 B
#define WS_RELB   131072           // f32  [4][49][49] = 38416 B

__device__ __forceinline__ v8f wmma_bf16(v16bf a, v16bf b, v8f c) {
  return __builtin_amdgcn_wmma_f32_16x16x32_bf16(false, a, false, b,
                                                 (short)0, c, false, false);
}

// 16-byte async copy global -> LDS (per-lane). Falls back to sync copy if the
// builtin is unavailable on this toolchain.
__device__ __forceinline__ void async_cp16(void* lds, const void* g) {
#if __has_builtin(__builtin_amdgcn_global_load_async_to_lds_b128)
  __builtin_amdgcn_global_load_async_to_lds_b128(
      (__attribute__((address_space(1))) v4i*)(void*)(const_cast<void*>(g)),
      (__attribute__((address_space(3))) v4i*)lds, 0, 0);
#else
  *(uint4*)lds = *(const uint4*)g;
#endif
}

__device__ __forceinline__ void async_wait0() {
#if __has_builtin(__builtin_amdgcn_s_wait_asynccnt)
  __builtin_amdgcn_s_wait_asynccnt(0);
#elif __has_builtin(__builtin_amdgcn_global_load_async_to_lds_b128)
  asm volatile("s_wait_asynccnt 0x0" ::: "memory");
#endif
}

// Load a 32-byte fragment as two 16-byte chunks (LDS or global).
__device__ __forceinline__ v16bf frag_ld2(const __bf16* p0, const __bf16* p1) {
  union { v16bf v; uint4 q[2]; } u;
  u.q[0] = *(const uint4*)p0;
  u.q[1] = *(const uint4*)p1;
  return u.v;
}

// Load 16 f32 (two runs of 8) and convert to a bf16 A-fragment.
__device__ __forceinline__ v16bf frag_ld_f32(const float* p0, const float* p1) {
  float4 a0 = *(const float4*)p0;
  float4 a1 = *(const float4*)(p0 + 4);
  float4 b0 = *(const float4*)p1;
  float4 b1 = *(const float4*)(p1 + 4);
  v16bf v;
  v[0]=(__bf16)a0.x; v[1]=(__bf16)a0.y; v[2]=(__bf16)a0.z; v[3]=(__bf16)a0.w;
  v[4]=(__bf16)a1.x; v[5]=(__bf16)a1.y; v[6]=(__bf16)a1.z; v[7]=(__bf16)a1.w;
  v[8]=(__bf16)b0.x; v[9]=(__bf16)b0.y; v[10]=(__bf16)b0.z; v[11]=(__bf16)b0.w;
  v[12]=(__bf16)b1.x; v[13]=(__bf16)b1.y; v[14]=(__bf16)b1.z; v[15]=(__bf16)b1.w;
  return v;
}

// ---------------------------------------------------------------------------
// Prep: transpose+convert weights to bf16, expand relative-position bias.
// ---------------------------------------------------------------------------
__global__ __launch_bounds__(256) void swin_prep_kernel(
    const float* __restrict__ qkv_w,    // [128][384]
    const float* __restrict__ proj_w,   // [128][128]
    const float* __restrict__ rel_t,    // [169][4]
    __bf16* __restrict__ wqkvT,         // [384][128]
    __bf16* __restrict__ wprojT,        // [128][128]
    float* __restrict__ relb)           // [4][49][49]
{
  int t = blockIdx.x * 256 + threadIdx.x;
  if (t < 384 * 128) {
    int n = t >> 7, k = t & 127;
    wqkvT[n * 128 + k] = (__bf16)qkv_w[k * 384 + n];
  } else if (t < 384 * 128 + 128 * 128) {
    int u = t - 384 * 128;
    int n = u >> 7, k = u & 127;
    wprojT[n * 128 + k] = (__bf16)proj_w[k * 128 + n];
  } else if (t < 384 * 128 + 128 * 128 + H_NUM * NT * NT) {
    int u = t - (384 * 128 + 128 * 128);
    int h = u / (NT * NT), rem = u % (NT * NT);
    int i = rem / NT, j = rem % NT;
    int yi = i / WW, xi = i % WW, yj = j / WW, xj = j % WW;
    int idx = (yi - yj + WH - 1) * (2 * WW - 1) + (xi - xj + WW - 1);
    relb[(h * NT + i) * NT + j] = rel_t[idx * H_NUM + h];
  }
}

// ---------------------------------------------------------------------------
// Fused window attention: one workgroup per window, 8 wave32 waves.
// ---------------------------------------------------------------------------
__global__ __launch_bounds__(256) void swin_attn_kernel(
    const float* __restrict__ x,       // [4096][49][128]
    const float* __restrict__ maskg,   // [64][49][49]
    const float* __restrict__ qkv_b,   // [384]
    const float* __restrict__ proj_b,  // [128]
    const __bf16* __restrict__ wqkvT,  // [384][128]
    const __bf16* __restrict__ wprojT, // [128][128]
    const float* __restrict__ relb,    // [4][49][49]
    float* __restrict__ out)           // [4096][49][128]
{
  extern __shared__ char smem[];
  __bf16* sQK  = (__bf16*)(smem + OFF_QK);   // [64][264] q cols 0..127, k 128..255
  __bf16* sWP  = (__bf16*)(smem + OFF_QK);   // [128][128] proj weights (phase D)
  __bf16* sVT  = (__bf16*)(smem + OFF_VT);   // [4][32][72] V transposed
  __bf16* sX   = (__bf16*)(smem + OFF_X);    // [64][136]
  __bf16* sO   = sX;                         // reused in phase C/D
  __bf16* sWT0 = (__bf16*)(smem + OFF_WT0);  // [384][48] weight panel buf 0
  __bf16* sWT1 = (__bf16*)(smem + OFF_WT1);  // [384][48] weight panel buf 1
  float*  sS   = (float*)(smem + OFF_S);     // [4][64][68] scores / probs

  const int tid  = threadIdx.x;
  const int wave = tid >> 5;
  const int lane = tid & 31;
  const int lm   = lane & 15;   // m (A) / n (B,C) position
  const int lh   = lane >> 4;   // half selector
  const int b    = blockIdx.x;

  // ---- issue async stage of weight panel 0 (double-buffered pipeline) ----
  {
    __bf16* buf = sWT0;
#pragma unroll
    for (int c0 = 0; c0 < 1536; c0 += 256) {
      int c = c0 + tid;
      int n = c >> 2, sub = (c & 3) << 3;         // 4 x 16B chunks per row
      async_cp16(buf + n * STR_WT + sub, wqkvT + n * 128 + sub);
    }
  }

  // ---- Load x window -> bf16 LDS, zero-pad rows 49..63 -------------------
  for (int idx = tid; idx < NP * (C_DIM / 4); idx += 256) {
    int row = idx >> 5;
    int c4  = (idx & 31) << 2;
    float4 f = make_float4(0.f, 0.f, 0.f, 0.f);
    if (row < NT) f = *(const float4*)(x + ((size_t)b * NT + row) * C_DIM + c4);
    union { __bf16 h[4]; uint2 u; } pk;
    pk.h[0] = (__bf16)f.x; pk.h[1] = (__bf16)f.y;
    pk.h[2] = (__bf16)f.z; pk.h[3] = (__bf16)f.w;
    *(uint2*)(sX + row * STR_X + c4) = pk.u;
  }

  // ---- Phase A: qkv = x @ qkv_w (+bias, q*scale) -------------------------
  // 4 row-tiles x 24 col-tiles; wave w owns col-tiles w*3..w*3+2.
  v8f accQ[12] = {};
#pragma unroll
  for (int kt = 0; kt < 4; ++kt) {
    __bf16* cur = (kt & 1) ? sWT1 : sWT0;
    async_wait0();          // this wave's staged panel is in LDS
    __syncthreads();        // every wave's chunks are visible
    if (kt < 3) {           // stage next panel into the other buffer
      __bf16* nxt = (kt & 1) ? sWT0 : sWT1;
#pragma unroll
      for (int c0 = 0; c0 < 1536; c0 += 256) {
        int c = c0 + tid;
        int n = c >> 2, sub = (c & 3) << 3;
        async_cp16(nxt + n * STR_WT + sub, wqkvT + n * 128 + (kt + 1) * 32 + sub);
      }
    }
#pragma unroll
    for (int rt = 0; rt < 4; ++rt) {
      const __bf16* pa = sX + (rt * 16 + lm) * STR_X + kt * 32 + 8 * lh;
      v16bf ax = frag_ld2(pa, pa + 16);
#pragma unroll
      for (int ci = 0; ci < 3; ++ci) {
        int ct = wave * 3 + ci;
        const __bf16* pb = cur + (ct * 16 + lm) * STR_WT + 8 * lh;
        v16bf bw = frag_ld2(pb, pb + 16);
        accQ[rt * 3 + ci] = wmma_bf16(ax, bw, accQ[rt * 3 + ci]);
      }
    }
    __syncthreads();        // all reads of 'cur' done before it is re-staged
  }
  // epilogue: bias, scale q, scatter into sQK / sVT
  const float scale = 0.17677669529663687f;  // 1/sqrt(32)
#pragma unroll
  for (int rt = 0; rt < 4; ++rt) {
#pragma unroll
    for (int ci = 0; ci < 3; ++ci) {
      int ct = wave * 3 + ci;
      int col = ct * 16 + lm;
      float bias = qkv_b[col];
#pragma unroll
      for (int r = 0; r < 8; ++r) {
        int row = rt * 16 + r + 8 * lh;
        float v = accQ[rt * 3 + ci][r] + bias;
        if (col < 128) {
          sQK[row * STR_QK + col] = (__bf16)(v * scale);       // q (scaled)
        } else if (col < 256) {
          sQK[row * STR_QK + col] = (__bf16)v;                 // k
        } else {
          int cv = col - 256;
          sVT[(cv >> 5) * (HD * STR_VT) + (cv & 31) * STR_VT + row] = (__bf16)v;
        }
      }
    }
  }
  __syncthreads();

  // ---- Phase B: scores S = q @ k^T + rel_bias + mask ---------------------
  {
    int h  = wave >> 1;
    int rh = wave & 1;
    const float* maskp = maskg + (size_t)(b & (NWIN - 1)) * NT * NT;
    const float* relh  = relb + h * NT * NT;
#pragma unroll
    for (int ri = 0; ri < 2; ++ri) {
      int rt = rh * 2 + ri;
      const __bf16* pa = sQK + (rt * 16 + lm) * STR_QK + h * HD + 8 * lh;
      v16bf aq = frag_ld2(pa, pa + 16);
#pragma unroll
      for (int ct = 0; ct < 4; ++ct) {
        const __bf16* pb = sQK + (ct * 16 + lm) * STR_QK + 128 + h * HD + 8 * lh;
        v16bf bk = frag_ld2(pb, pb + 16);
        v8f z = {};
        v8f cc = wmma_bf16(aq, bk, z);
        int j = ct * 16 + lm;
#pragma unroll
        for (int r = 0; r < 8; ++r) {
          int i = rt * 16 + r + 8 * lh;
          float s = -1.0e30f;
          if (i < NT && j < NT)
            s = cc[r] + relh[i * NT + j] + maskp[i * NT + j];
          sS[(h * NP + i) * STR_S + j] = s;
        }
      }
    }
  }
  __syncthreads();  // sQK reads complete; region becomes free for sWP

  // ---- prefetch proj weights into the dead sQK region (async, overlapped)
#pragma unroll
  for (int c0 = 0; c0 < 2048; c0 += 256) {
    int c = c0 + tid;                     // 2048 x 16B chunks = 32 KB
    async_cp16(sWP + c * 8, wprojT + c * 8);
  }

  // ---- softmax over j: one (head,row) per thread -------------------------
  {
    int h = tid >> 6, i = tid & 63;
    float* row = sS + (h * NP + i) * STR_S;
    float mx = -1.0e30f;
    for (int j = 0; j < NP; ++j) mx = fmaxf(mx, row[j]);
    float sum = 0.f;
    for (int j = 0; j < NP; ++j) { float e = __expf(row[j] - mx); row[j] = e; sum += e; }
    float inv = 1.0f / sum;
    for (int j = 0; j < NP; ++j) row[j] *= inv;
  }
  __syncthreads();

  // ---- Phase C: O = P @ V ------------------------------------------------
  {
    int h  = wave >> 1;
    int rh = wave & 1;
    v8f accO[2][2] = {};
#pragma unroll
    for (int kt = 0; kt < 2; ++kt) {
#pragma unroll
      for (int ri = 0; ri < 2; ++ri) {
        int rt = rh * 2 + ri;
        const float* pa = sS + (h * NP + rt * 16 + lm) * STR_S + kt * 32 + 8 * lh;
        v16bf ap = frag_ld_f32(pa, pa + 16);
#pragma unroll
        for (int ct = 0; ct < 2; ++ct) {
          const __bf16* pb = sVT + h * (HD * STR_VT) + (ct * 16 + lm) * STR_VT
                             + kt * 32 + 8 * lh;
          v16bf bv = frag_ld2(pb, pb + 16);
          accO[ri][ct] = wmma_bf16(ap, bv, accO[ri][ct]);
        }
      }
    }
#pragma unroll
    for (int ri = 0; ri < 2; ++ri) {
      int rt = rh * 2 + ri;
#pragma unroll
      for (int ct = 0; ct < 2; ++ct) {
        int col = h * HD + ct * 16 + lm;
#pragma unroll
        for (int r = 0; r < 8; ++r) {
          int row = rt * 16 + r + 8 * lh;
          sO[row * STR_X + col] = (__bf16)accO[ri][ct][r];
        }
      }
    }
  }
  async_wait0();    // proj weights staged
  __syncthreads();

  // ---- Phase D: out = O @ proj_w + proj_b (B frags from LDS) -------------
  {
    int rt = wave >> 1;
    int cb = (wave & 1) * 4;
    v8f accP[4] = {};
#pragma unroll
    for (int kt = 0; kt < 4; ++kt) {
      const __bf16* pa = sO + (rt * 16 + lm) * STR_X + kt * 32 + 8 * lh;
      v16bf ao = frag_ld2(pa, pa + 16);
#pragma unroll
      for (int ci = 0; ci < 4; ++ci) {
        int ct = cb + ci;
        const __bf16* pb = sWP + (ct * 16 + lm) * 128 + kt * 32 + 8 * lh;
        v16bf bw = frag_ld2(pb, pb + 16);
        accP[ci] = wmma_bf16(ao, bw, accP[ci]);
      }
    }
#pragma unroll
    for (int ci = 0; ci < 4; ++ci) {
      int col = (cb + ci) * 16 + lm;
      float bias = proj_b[col];
#pragma unroll
      for (int r = 0; r < 8; ++r) {
        int row = rt * 16 + r + 8 * lh;
        if (row < NT)
          out[((size_t)b * NT + row) * C_DIM + col] = accP[ci][r] + bias;
      }
    }
  }
}

// ---------------------------------------------------------------------------
extern "C" void kernel_launch(void* const* d_in, const int* in_sizes, int n_in,
                              void* d_out, int out_size, void* d_ws, size_t ws_size,
                              hipStream_t stream) {
  const float* x      = (const float*)d_in[0];
  const float* mask   = (const float*)d_in[1];
  const float* qkv_w  = (const float*)d_in[2];
  const float* qkv_b  = (const float*)d_in[3];
  const float* proj_w = (const float*)d_in[4];
  const float* proj_b = (const float*)d_in[5];
  const float* rel_t  = (const float*)d_in[6];
  float* out = (float*)d_out;

  char* ws = (char*)d_ws;
  __bf16* wqkvT  = (__bf16*)(ws + WS_WQKVT);
  __bf16* wprojT = (__bf16*)(ws + WS_WPROJT);
  float*  relb   = (float*)(ws + WS_RELB);

  (void)hipFuncSetAttribute((const void*)swin_attn_kernel,
                            hipFuncAttributeMaxDynamicSharedMemorySize,
                            SMEM_BYTES);

  int prep_threads = 384 * 128 + 128 * 128 + H_NUM * NT * NT;  // 75140
  swin_prep_kernel<<<(prep_threads + 255) / 256, 256, 0, stream>>>(
      qkv_w, proj_w, rel_t, wqkvT, wprojT, relb);

  swin_attn_kernel<<<B_TOT, 256, SMEM_BYTES, stream>>>(
      x, mask, qkv_b, proj_b, wqkvT, wprojT, relb, out);
}
